// QGcnn_43911745634382
// MI455X (gfx1250) — matrix operand order repaired and checked by reading, not compiled, not run on hardware
//
#include <hip/hip_runtime.h>

#define N_NODES 50000
#define N_EDGES 800000
#define E2      (2 * N_EDGES)
#define LDA     136   // 128 halfs + 8 pad, keeps 16B alignment of fragment chunks

typedef __attribute__((ext_vector_type(16))) _Float16 v16h;
typedef __attribute__((ext_vector_type(8)))  _Float16 v8h;
typedef __attribute__((ext_vector_type(8)))  float    v8f;

// ---------------------------------------------------------------------------
// WMMA helpers (CDNA5 gfx1250, wave32)
// ---------------------------------------------------------------------------
__device__ __forceinline__ v8f wmma_f16(v16h a, v16h b, v8f c) {
  // D = A(16x32 f16) * B(32x16 f16) + C(16x16 f32)
  return __builtin_amdgcn_wmma_f32_16x16x32_f16(false, a, false, b, (short)0, c,
                                                false, false);
}

// A fragment from a row-major f16 tile in LDS (16 rows x 128 cols, stride LDA).
// ISA 16-bit A 16x32 layout: lane L -> row L%16; halfs[0..7]=K kb..kb+7,
// halfs[8..15]=K kb+16..kb+23 with kb=(L/16)*8.
__device__ __forceinline__ v16h load_a_frag(const _Float16* tile, int lane, int kt) {
  const int r  = lane & 15;
  const int kb = (lane >> 4) << 3;
  const _Float16* p = tile + r * LDA + kt * 32 + kb;
  v8h lo = *(const v8h*)p;
  v8h hi = *(const v8h*)(p + 16);
  return __builtin_shufflevector(lo, hi, 0, 1, 2, 3, 4, 5, 6, 7,
                                 8, 9, 10, 11, 12, 13, 14, 15);
}

// B fragment from pre-packed weights: per (nt,kt) tile, lane L holds 16
// contiguous halfs = column nt*16+L%16, K rows kt*32+(L/16)*8+{0..7, 16..23}.
__device__ __forceinline__ v16h load_b_frag(const _Float16* packW, int lane,
                                            int nt, int kt, int KT) {
  return *(const v16h*)(packW + (size_t)(((nt * KT + kt) * 32 + lane) * 16));
}

// One edge-feature half-row (64 f16) into an LDS A-tile row: 8x b128.
__device__ __forceinline__ void build_a_row(const _Float16* __restrict__ x, int nid,
                                            _Float16* dp) {
  const v8h* xr = (const v8h*)(x + (size_t)nid * 64);
#pragma unroll
  for (int i = 0; i < 8; ++i) *(v8h*)(dp + i * 8) = xr[i];
}

// ---------------------------------------------------------------------------
// Weight packing: f32 (128 x N, row-major) -> f16 WMMA-B tiles, N padded.
// total = Npad*128 halfs; tile = 512 halfs.
// ---------------------------------------------------------------------------
__global__ void pack_w_kernel(const float* __restrict__ W, _Float16* __restrict__ out,
                              int N, int total) {
  int t = blockIdx.x * blockDim.x + threadIdx.x;
  if (t >= total) return;
  const int KT = 4;                 // K = 128 always
  int tileId = t >> 9;
  int within = t & 511;
  int lane = within >> 4;
  int j    = within & 15;
  int kt = tileId % KT;
  int nt = tileId / KT;
  int k = kt * 32 + ((j >> 3) << 4) + ((lane >> 4) << 3) + (j & 7);
  int n = nt * 16 + (lane & 15);
  float v = (n < N) ? W[k * N + n] : 0.0f;
  out[t] = (_Float16)v;
}

__global__ void zero_f32(float* __restrict__ p, int n) {
  int i = blockIdx.x * blockDim.x + threadIdx.x;
  if (i < n) p[i] = 0.0f;
}

__global__ void cvt_f32_f16(const float* __restrict__ in, _Float16* __restrict__ out,
                            int n) {
  int i = blockIdx.x * blockDim.x + threadIdx.x;
  if (i < n) out[i] = (_Float16)in[i];
}

__global__ void count_kernel(const int* __restrict__ dstArr, float* __restrict__ cnt,
                             int e2) {
  int i = blockIdx.x * blockDim.x + threadIdx.x;
  if (i < e2) atomicAdd(&cnt[dstArr[i]], 1.0f);
}

// agg(f32) / max(cnt,1) -> f16 node features for the next layer's A-tiles
__global__ void finalize_nodes(const float* __restrict__ agg,
                               const float* __restrict__ cnt,
                               _Float16* __restrict__ xout, int n) {
  int i = blockIdx.x * blockDim.x + threadIdx.x;
  if (i < n) xout[i] = (_Float16)(agg[i] / fmaxf(cnt[i >> 6], 1.0f));
}

// single-block sum of n partials -> out[slot]
__global__ void reduce_sum_kernel(const float* __restrict__ p, int n,
                                  float* __restrict__ out, int slot) {
  __shared__ float sred[16];
  float s = 0.0f;
  for (int i = threadIdx.x; i < n; i += blockDim.x) s += p[i];
#pragma unroll
  for (int off = 16; off > 0; off >>= 1) s += __shfl_xor(s, off, 32);
  int wave = threadIdx.x >> 5, lane = threadIdx.x & 31;
  if (lane == 0) sred[wave] = s;
  __syncthreads();
  if (threadIdx.x == 0) {
    float t = 0.0f;
    for (int w = 0; w < (int)(blockDim.x >> 5); ++w) t += sred[w];
    out[slot] = t;
  }
}

__global__ void loss_final_kernel(const float* __restrict__ lossAcc,
                                  float* __restrict__ out) {
  float sl1 = lossAcc[0] / ((float)N_EDGES * 128.0f);
  float sl2 = lossAcc[1] / ((float)N_EDGES * 3.0f);
  out[(size_t)N_EDGES * 3] = 0.5f * (sl1 + sl2);
}

// ---------------------------------------------------------------------------
// node_conv: per-wave 16-edge tile; MLP 128->128(leaky)->64; atomic scatter.
// grid: E2/16 tiles, 4 waves/block.
// ---------------------------------------------------------------------------
__global__ __launch_bounds__(128) void node_conv_kernel(
    const _Float16* __restrict__ x,
    const int* __restrict__ srcArr, const int* __restrict__ dstArr,
    const _Float16* __restrict__ W0p, const float* __restrict__ b0,
    const _Float16* __restrict__ W1p, const float* __restrict__ b1,
    float* __restrict__ agg) {
  __shared__ __align__(16) _Float16 sA[4][16 * LDA];
  __shared__ __align__(16) _Float16 sH[4][16 * LDA];
  __shared__ int sDst[4][16];
  const int wave = threadIdx.x >> 5;
  const int lane = threadIdx.x & 31;
  const int tile = blockIdx.x * 4 + wave;
  const long base = (long)tile * 16;
  _Float16* A = sA[wave];
  _Float16* H = sH[wave];

  if (lane < 16) sDst[wave][lane] = dstArr[base + lane];
  {
    int row = lane >> 1, hf = lane & 1;
    long e = base + row;
    int nid = hf ? dstArr[e] : srcArr[e];
    build_a_row(x, nid, A + row * LDA + hf * 64);
  }
  __syncthreads();

  const int col0 = lane & 15;
  const int rOff = (lane >> 4) << 3;

  v8f acc[8];
#pragma unroll
  for (int nt = 0; nt < 8; ++nt) {
    float bv = b0[nt * 16 + col0];
#pragma unroll
    for (int j = 0; j < 8; ++j) acc[nt][j] = bv;
  }
#pragma unroll
  for (int kt = 0; kt < 4; ++kt) {
    v16h a = load_a_frag(A, lane, kt);
#pragma unroll
    for (int nt = 0; nt < 8; ++nt)
      acc[nt] = wmma_f16(a, load_b_frag(W0p, lane, nt, kt, 4), acc[nt]);
  }
#pragma unroll
  for (int nt = 0; nt < 8; ++nt)
#pragma unroll
    for (int j = 0; j < 8; ++j) {
      float v = acc[nt][j];
      v = v > 0.0f ? v : 0.01f * v;       // leaky relu
      H[(j + rOff) * LDA + nt * 16 + col0] = (_Float16)v;
    }
  __syncthreads();

  v8f acc2[4];
#pragma unroll
  for (int nt = 0; nt < 4; ++nt) {
    float bv = b1[nt * 16 + col0];
#pragma unroll
    for (int j = 0; j < 8; ++j) acc2[nt][j] = bv;
  }
#pragma unroll
  for (int kt = 0; kt < 4; ++kt) {
    v16h a = load_a_frag(H, lane, kt);
#pragma unroll
    for (int nt = 0; nt < 4; ++nt)
      acc2[nt] = wmma_f16(a, load_b_frag(W1p, lane, nt, kt, 4), acc2[nt]);
  }

  int drow[8];
#pragma unroll
  for (int j = 0; j < 8; ++j) drow[j] = sDst[wave][j + rOff];
#pragma unroll
  for (int nt = 0; nt < 4; ++nt)
#pragma unroll
    for (int j = 0; j < 8; ++j)
      atomicAdd(&agg[(size_t)drow[j] * 64 + nt * 16 + col0], acc2[nt][j]);
}

// ---------------------------------------------------------------------------
// edge_conv1: 16 edge-pairs per wave via 2 passes (rows 0-7 = edges e..e+7,
// rows 8-15 = mirrored e+N..). Pair combine via shfl_xor(16). Then fused
// 130->128 linear: WMMA over K=128 + rank-2 VALU fixup (actions/angles).
// ---------------------------------------------------------------------------
__global__ __launch_bounds__(128) void edge_conv1_kernel(
    const _Float16* __restrict__ x,
    const int* __restrict__ srcArr, const int* __restrict__ dstArr,
    const _Float16* __restrict__ W0p, const float* __restrict__ b0,
    const _Float16* __restrict__ W1p, const float* __restrict__ b1,
    const _Float16* __restrict__ Wlp, const float* __restrict__ Wl,
    const float* __restrict__ bl,
    const float* __restrict__ actions, const float* __restrict__ angles,
    _Float16* __restrict__ feOut, float* __restrict__ lossPartial) {
  __shared__ __align__(16) _Float16 sA[4][16 * LDA];
  __shared__ __align__(16) _Float16 sH[4][16 * LDA];
  __shared__ __align__(16) _Float16 sFe[4][16 * LDA];
  __shared__ float sAct[4][16], sAng[4][16];
  const int wave = threadIdx.x >> 5;
  const int lane = threadIdx.x & 31;
  const int tile = blockIdx.x * 4 + wave;
  const long base = (long)tile * 16;
  _Float16* A = sA[wave];
  _Float16* H = sH[wave];
  _Float16* Fe = sFe[wave];
  const int col0 = lane & 15;
  const int rOff = (lane >> 4) << 3;

  if (lane < 16) {
    sAct[wave][lane] = actions[base + lane];
    sAng[wave][lane] = angles[base + lane];
  }
  float lsum = 0.0f;

  for (int pass = 0; pass < 2; ++pass) {
    {
      int row = lane >> 1, hf = lane & 1;
      long e = base + pass * 8 + (row & 7);
      if (row >= 8) e += N_EDGES;       // mirrored copy (src/dst swapped)
      int nid = hf ? dstArr[e] : srcArr[e];
      build_a_row(x, nid, A + row * LDA + hf * 64);
    }
    __syncthreads();

    v8f acc[8];
#pragma unroll
    for (int nt = 0; nt < 8; ++nt) {
      float bv = b0[nt * 16 + col0];
#pragma unroll
      for (int j = 0; j < 8; ++j) acc[nt][j] = bv;
    }
#pragma unroll
    for (int kt = 0; kt < 4; ++kt) {
      v16h a = load_a_frag(A, lane, kt);
#pragma unroll
      for (int nt = 0; nt < 8; ++nt)
        acc[nt] = wmma_f16(a, load_b_frag(W0p, lane, nt, kt, 4), acc[nt]);
    }
#pragma unroll
    for (int nt = 0; nt < 8; ++nt)
#pragma unroll
      for (int j = 0; j < 8; ++j) {
        float v = acc[nt][j];
        v = v > 0.0f ? v : 0.01f * v;
        H[(j + rOff) * LDA + nt * 16 + col0] = (_Float16)v;
      }
    __syncthreads();

    v8f acc2[8];
#pragma unroll
    for (int nt = 0; nt < 8; ++nt) {
      float bv = b1[nt * 16 + col0];
#pragma unroll
      for (int j = 0; j < 8; ++j) acc2[nt][j] = bv;
    }
#pragma unroll
    for (int kt = 0; kt < 4; ++kt) {
      v16h a = load_a_frag(H, lane, kt);
#pragma unroll
      for (int nt = 0; nt < 8; ++nt)
        acc2[nt] = wmma_f16(a, load_b_frag(W1p, lane, nt, kt, 4), acc2[nt]);
    }

    // pair combine: lanes L / L+16 hold fe1 / fe2 of the same edge
#pragma unroll
    for (int nt = 0; nt < 8; ++nt)
#pragma unroll
      for (int j = 0; j < 8; ++j) {
        float v = acc2[nt][j];
        float p = __shfl_xor(v, 16, 32);
        if (lane < 16) {
          float d = v - p;
          lsum += d * d;
          Fe[(pass * 8 + j) * LDA + nt * 16 + lane] = (_Float16)(0.5f * (v + p));
        }
      }
    __syncthreads();
  }

  // out = [fe, actions, angles] @ Wl + bl   (K=128 via WMMA + 2-row fixup)
  v8f accL[8];
#pragma unroll
  for (int nt = 0; nt < 8; ++nt) {
    float bv = bl[nt * 16 + col0];
#pragma unroll
    for (int j = 0; j < 8; ++j) accL[nt][j] = bv;
  }
#pragma unroll
  for (int kt = 0; kt < 4; ++kt) {
    v16h a = load_a_frag(Fe, lane, kt);
#pragma unroll
    for (int nt = 0; nt < 8; ++nt)
      accL[nt] = wmma_f16(a, load_b_frag(Wlp, lane, nt, kt, 4), accL[nt]);
  }
#pragma unroll
  for (int nt = 0; nt < 8; ++nt)
#pragma unroll
    for (int j = 0; j < 8; ++j) {
      int row = j + rOff;
      int n = nt * 16 + col0;
      float v = accL[nt][j] + sAct[wave][row] * Wl[128 * 128 + n]
                            + sAng[wave][row] * Wl[129 * 128 + n];
      feOut[(size_t)(base + row) * 128 + n] = (_Float16)v;
    }

#pragma unroll
  for (int off = 16; off > 0; off >>= 1) lsum += __shfl_xor(lsum, off, 32);
  if (lane == 0) lossPartial[tile] = lsum;
}

// ---------------------------------------------------------------------------
// edge_conv2: MLP 128->128(leaky)->3 (N padded to 16), pair combine,
// then scalar 131->3 linear with init_fe = ec1 output.
// ---------------------------------------------------------------------------
__global__ __launch_bounds__(128) void edge_conv2_kernel(
    const _Float16* __restrict__ x,
    const int* __restrict__ srcArr, const int* __restrict__ dstArr,
    const _Float16* __restrict__ W0p, const float* __restrict__ b0,
    const _Float16* __restrict__ W1p, const float* __restrict__ b1,
    const float* __restrict__ Wl2, const float* __restrict__ bl2,
    const _Float16* __restrict__ fe1out,
    float* __restrict__ outFe, float* __restrict__ lossPartial) {
  __shared__ __align__(16) _Float16 sA[4][16 * LDA];
  __shared__ __align__(16) _Float16 sH[4][16 * LDA];
  __shared__ __align__(16) _Float16 sFe[4][16 * 16];
  const int wave = threadIdx.x >> 5;
  const int lane = threadIdx.x & 31;
  const int tile = blockIdx.x * 4 + wave;
  const long base = (long)tile * 16;
  _Float16* A = sA[wave];
  _Float16* H = sH[wave];
  const int col0 = lane & 15;
  const int rOff = (lane >> 4) << 3;
  float lsum = 0.0f;

  for (int pass = 0; pass < 2; ++pass) {
    {
      int row = lane >> 1, hf = lane & 1;
      long e = base + pass * 8 + (row & 7);
      if (row >= 8) e += N_EDGES;
      int nid = hf ? dstArr[e] : srcArr[e];
      build_a_row(x, nid, A + row * LDA + hf * 64);
    }
    __syncthreads();

    v8f acc[8];
#pragma unroll
    for (int nt = 0; nt < 8; ++nt) {
      float bv = b0[nt * 16 + col0];
#pragma unroll
      for (int j = 0; j < 8; ++j) acc[nt][j] = bv;
    }
#pragma unroll
    for (int kt = 0; kt < 4; ++kt) {
      v16h a = load_a_frag(A, lane, kt);
#pragma unroll
      for (int nt = 0; nt < 8; ++nt)
        acc[nt] = wmma_f16(a, load_b_frag(W0p, lane, nt, kt, 4), acc[nt]);
    }
#pragma unroll
    for (int nt = 0; nt < 8; ++nt)
#pragma unroll
      for (int j = 0; j < 8; ++j) {
        float v = acc[nt][j];
        v = v > 0.0f ? v : 0.01f * v;
        H[(j + rOff) * LDA + nt * 16 + col0] = (_Float16)v;
      }
    __syncthreads();

    v8f acc2;
    {
      float bv = (col0 < 3) ? b1[col0] : 0.0f;
#pragma unroll
      for (int j = 0; j < 8; ++j) acc2[j] = bv;
    }
#pragma unroll
    for (int kt = 0; kt < 4; ++kt)
      acc2 = wmma_f16(load_a_frag(H, lane, kt),
                      load_b_frag(W1p, lane, 0, kt, 4), acc2);

#pragma unroll
    for (int j = 0; j < 8; ++j) {
      float v = acc2[j];
      float p = __shfl_xor(v, 16, 32);
      if (lane < 16) {
        if (col0 < 3) {
          float d = v - p;
          lsum += d * d;
        }
        sFe[wave][(pass * 8 + j) * 16 + lane] = (_Float16)(0.5f * (v + p));
      }
    }
    __syncthreads();
  }

  // out = [fe(3), init_fe(128)] @ Wl2 + bl2  (tiny: scalar per edge)
  if (lane < 16) {
    long e = base + lane;
    float o0 = bl2[0], o1 = bl2[1], o2 = bl2[2];
#pragma unroll
    for (int k = 0; k < 3; ++k) {
      float f = (float)sFe[wave][lane * 16 + k];
      o0 += f * Wl2[k * 3 + 0];
      o1 += f * Wl2[k * 3 + 1];
      o2 += f * Wl2[k * 3 + 2];
    }
    const _Float16* fr = fe1out + (size_t)e * 128;
    for (int k = 0; k < 128; ++k) {
      float f = (float)fr[k];
      const float* wr = &Wl2[(3 + k) * 3];
      o0 += f * wr[0];
      o1 += f * wr[1];
      o2 += f * wr[2];
    }
    outFe[e * 3 + 0] = o0;
    outFe[e * 3 + 1] = o1;
    outFe[e * 3 + 2] = o2;
  }

#pragma unroll
  for (int off = 16; off > 0; off >>= 1) lsum += __shfl_xor(lsum, off, 32);
  if (lane == 0) lossPartial[tile] = lsum;
}

// ---------------------------------------------------------------------------
extern "C" void kernel_launch(void* const* d_in, const int* in_sizes, int n_in,
                              void* d_out, int out_size, void* d_ws, size_t ws_size,
                              hipStream_t stream) {
  (void)in_sizes; (void)n_in; (void)out_size; (void)ws_size;
  const float* node_features = (const float*)d_in[0];
  const int*   edge_index    = (const int*)d_in[1];
  const float* angles        = (const float*)d_in[2];
  const float* actions       = (const float*)d_in[3];
  // d_in[4] = gt_edges (unused by reference)
  const float* nc1_W0 = (const float*)d_in[5];
  const float* nc1_b0 = (const float*)d_in[6];
  const float* nc1_W1 = (const float*)d_in[7];
  const float* nc1_b1 = (const float*)d_in[8];
  const float* ec1_W0 = (const float*)d_in[9];
  const float* ec1_b0 = (const float*)d_in[10];
  const float* ec1_W1 = (const float*)d_in[11];
  const float* ec1_b1 = (const float*)d_in[12];
  const float* ec1_Wl = (const float*)d_in[13];
  const float* ec1_bl = (const float*)d_in[14];
  const float* nc2_W0 = (const float*)d_in[15];
  const float* nc2_b0 = (const float*)d_in[16];
  const float* nc2_W1 = (const float*)d_in[17];
  const float* nc2_b1 = (const float*)d_in[18];
  const float* ec2_W0 = (const float*)d_in[19];
  const float* ec2_b0 = (const float*)d_in[20];
  const float* ec2_W1 = (const float*)d_in[21];
  const float* ec2_b1 = (const float*)d_in[22];
  const float* ec2_Wl = (const float*)d_in[23];
  const float* ec2_bl = (const float*)d_in[24];

  const int* srcArr = edge_index;
  const int* dstArr = edge_index + E2;
  const int NTILES_E = N_EDGES / 16;   // 50000

  char* ws = (char*)d_ws;
  auto alloc = [&](size_t bytes) {
    char* p = ws;
    ws += (bytes + 255) & ~(size_t)255;
    return p;
  };
  float*    agg     = (float*)alloc((size_t)N_NODES * 64 * sizeof(float));
  float*    cnt     = (float*)alloc((size_t)N_NODES * sizeof(float));
  float*    lossAcc = (float*)alloc(256);
  float*    lossP1  = (float*)alloc((size_t)NTILES_E * sizeof(float));
  float*    lossP2  = (float*)alloc((size_t)NTILES_E * sizeof(float));
  _Float16* nfh     = (_Float16*)alloc((size_t)N_NODES * 64 * 2);
  _Float16* x1h     = (_Float16*)alloc((size_t)N_NODES * 64 * 2);
  _Float16* x2h     = (_Float16*)alloc((size_t)N_NODES * 64 * 2);
  _Float16* nc1W0p  = (_Float16*)alloc(16384 * 2);
  _Float16* nc1W1p  = (_Float16*)alloc(8192 * 2);
  _Float16* ec1W0p  = (_Float16*)alloc(16384 * 2);
  _Float16* ec1W1p  = (_Float16*)alloc(16384 * 2);
  _Float16* ec1Wlp  = (_Float16*)alloc(16384 * 2);
  _Float16* nc2W0p  = (_Float16*)alloc(16384 * 2);
  _Float16* nc2W1p  = (_Float16*)alloc(8192 * 2);
  _Float16* ec2W0p  = (_Float16*)alloc(16384 * 2);
  _Float16* ec2W1p  = (_Float16*)alloc(2048 * 2);
  _Float16* fe1out  = (_Float16*)alloc((size_t)N_EDGES * 128 * 2);

  // init / convert / pack
  zero_f32<<<(N_NODES * 64 + 255) / 256, 256, 0, stream>>>(agg, N_NODES * 64);
  zero_f32<<<(N_NODES + 255) / 256, 256, 0, stream>>>(cnt, N_NODES);
  cvt_f32_f16<<<(N_NODES * 64 + 255) / 256, 256, 0, stream>>>(node_features, nfh,
                                                              N_NODES * 64);
  pack_w_kernel<<<64, 256, 0, stream>>>(nc1_W0, nc1W0p, 128, 16384);
  pack_w_kernel<<<32, 256, 0, stream>>>(nc1_W1, nc1W1p, 64, 8192);
  pack_w_kernel<<<64, 256, 0, stream>>>(ec1_W0, ec1W0p, 128, 16384);
  pack_w_kernel<<<64, 256, 0, stream>>>(ec1_W1, ec1W1p, 128, 16384);
  pack_w_kernel<<<64, 256, 0, stream>>>(ec1_Wl, ec1Wlp, 128, 16384);  // rows 0..127 of 130x128
  pack_w_kernel<<<64, 256, 0, stream>>>(nc2_W0, nc2W0p, 128, 16384);
  pack_w_kernel<<<32, 256, 0, stream>>>(nc2_W1, nc2W1p, 64, 8192);
  pack_w_kernel<<<64, 256, 0, stream>>>(ec2_W0, ec2W0p, 128, 16384);
  pack_w_kernel<<<8, 256, 0, stream>>>(ec2_W1, ec2W1p, 3, 2048);      // pad N 3->16
  count_kernel<<<(E2 + 255) / 256, 256, 0, stream>>>(dstArr, cnt, E2);

  // node_conv 1
  node_conv_kernel<<<E2 / 16 / 4, 128, 0, stream>>>(
      nfh, srcArr, dstArr, nc1W0p, nc1_b0, nc1W1p, nc1_b1, agg);
  finalize_nodes<<<(N_NODES * 64 + 255) / 256, 256, 0, stream>>>(agg, cnt, x1h,
                                                                 N_NODES * 64);
  // edge_conv 1
  edge_conv1_kernel<<<NTILES_E / 4, 128, 0, stream>>>(
      x1h, srcArr, dstArr, ec1W0p, ec1_b0, ec1W1p, ec1_b1, ec1Wlp, ec1_Wl,
      ec1_bl, actions, angles, fe1out, lossP1);
  // node_conv 2
  zero_f32<<<(N_NODES * 64 + 255) / 256, 256, 0, stream>>>(agg, N_NODES * 64);
  node_conv_kernel<<<E2 / 16 / 4, 128, 0, stream>>>(
      x1h, srcArr, dstArr, nc2W0p, nc2_b0, nc2W1p, nc2_b1, agg);
  finalize_nodes<<<(N_NODES * 64 + 255) / 256, 256, 0, stream>>>(agg, cnt, x2h,
                                                                 N_NODES * 64);
  // edge_conv 2 -> final fe
  edge_conv2_kernel<<<NTILES_E / 4, 128, 0, stream>>>(
      x2h, srcArr, dstArr, ec2W0p, ec2_b0, ec2W1p, ec2_b1, ec2_Wl, ec2_bl,
      fe1out, (float*)d_out, lossP2);
  // loss reduction (contention-free, deterministic)
  reduce_sum_kernel<<<1, 512, 0, stream>>>(lossP1, NTILES_E, lossAcc, 0);
  reduce_sum_kernel<<<1, 512, 0, stream>>>(lossP2, NTILES_E, lossAcc, 1);
  loss_final_kernel<<<1, 1, 0, stream>>>(lossAcc, (float*)d_out);
}
